// GAT_MultiHeads_22746146800248
// MI455X (gfx1250) — matrix operand order repaired and checked by reading, not compiled, not run on hardware
//
#include <hip/hip_runtime.h>
#include <hip/hip_bf16.h>

typedef __attribute__((ext_vector_type(16))) _Float16 v16h;
typedef __attribute__((ext_vector_type(8)))  float    v8f;

#define NN     4096
#define F_IN   256
#define HEADS  8
#define DH     16
#define HD     128
#define NEG_SLOPE 0.3f
#define JT     128          // number of 32-wide j tiles (4096/32)

__device__ __forceinline__ float lrelu(float x) { return fmaxf(x, NEG_SLOPE * x); }

__device__ __forceinline__ v8f wmma_f16(v16h a, v16h b, v8f c) {
    return __builtin_amdgcn_wmma_f32_16x16x32_f16(false, a, false, b, (short)0, c, false, false);
}

// A-fragment (16x32, f16) from row-major fp32 matrix.
// lane L: M = L&15; element i: K = k0 + (L&16?8:0) + (i<8 ? i : 16+(i-8))
__device__ __forceinline__ v16h load_afrag_f32(const float* __restrict__ A, int lda,
                                               int row, int k0, int lane) {
    int base = k0 + ((lane & 16) ? 8 : 0);
    const float* p = A + (size_t)row * lda + base;
    float4 x0 = *(const float4*)(p + 0);
    float4 x1 = *(const float4*)(p + 4);
    float4 y0 = *(const float4*)(p + 16);
    float4 y1 = *(const float4*)(p + 20);
    v16h a;
    a[0]=(_Float16)x0.x;  a[1]=(_Float16)x0.y;  a[2]=(_Float16)x0.z;  a[3]=(_Float16)x0.w;
    a[4]=(_Float16)x1.x;  a[5]=(_Float16)x1.y;  a[6]=(_Float16)x1.z;  a[7]=(_Float16)x1.w;
    a[8]=(_Float16)y0.x;  a[9]=(_Float16)y0.y;  a[10]=(_Float16)y0.z; a[11]=(_Float16)y0.w;
    a[12]=(_Float16)y1.x; a[13]=(_Float16)y1.y; a[14]=(_Float16)y1.z; a[15]=(_Float16)y1.w;
    return a;
}

// ---------------- pack a row-major fp32 [K x 128] weight into f16 B-fragment layout ----------------
// dst frag index: ((kt*8 + nt)*32 + lane)*16 + i  <=  src[(kt*32 + (lane&16?8:0) + (i<8?i:i+8))*128 + nt*16 + (lane&15)]
__global__ void k_packB(const float* __restrict__ src, _Float16* __restrict__ dst, int KT) {
    int tid = blockIdx.x * blockDim.x + threadIdx.x;     // KT*8*32 threads
    int lane = tid & 31;
    int w = tid >> 5;
    int nt = w & 7;
    int kt = w >> 3;
    if (kt >= KT) return;
    int krow = kt * 32 + ((lane & 16) ? 8 : 0);
    int col = nt * 16 + (lane & 15);
    _Float16* d = dst + (((size_t)(kt * 8 + nt) * 32 + lane) << 4);
#pragma unroll
    for (int i = 0; i < 16; ++i)
        d[i] = (_Float16)src[(size_t)(krow + (i < 8 ? i : i + 8)) * HD + col];
}

// ---------------- K1: score = inputs @ W_w + W_b  ([4096,256]x[256,128]) ----------------
__global__ void k_score_gemm(const float* __restrict__ inputs, const _Float16* __restrict__ packW,
                             const float* __restrict__ Wb, float* __restrict__ score) {
    int wave = threadIdx.x >> 5, lane = threadIdx.x & 31;
    int tile = blockIdx.x * 4 + wave;            // 2048 tiles
    if (tile >= (NN / 16) * (HD / 16)) return;
    int tm = tile >> 3, tn = tile & 7;
    int row = tm * 16 + (lane & 15);
    int col = tn * 16 + (lane & 15);
    const v16h* bp = (const v16h*)packW;
    v8f acc = {0.f,0.f,0.f,0.f,0.f,0.f,0.f,0.f};
#pragma unroll
    for (int kt = 0; kt < F_IN / 32; ++kt)
        acc = wmma_f16(load_afrag_f32(inputs, F_IN, row, kt * 32, lane),
                       bp[(size_t)(kt * 8 + tn) * 32 + lane], acc);
    float bv = Wb[col];
    int rbase = tm * 16 + ((lane & 16) ? 8 : 0);
#pragma unroll
    for (int r = 0; r < 8; ++r)
        score[(size_t)(rbase + r) * HD + col] = acc[r] + bv;
}

// ---------------- K2: f1[h][j], f1t[j][h], f2[h][j] ----------------
__global__ void k_f1f2(const float* __restrict__ score, const float* __restrict__ wu,
                       const float* __restrict__ wv, float* __restrict__ f1,
                       float* __restrict__ f1t, float* __restrict__ f2) {
    int j = blockIdx.x * blockDim.x + threadIdx.x;
    if (j >= NN) return;
    const float* srow = score + (size_t)j * HD;
#pragma unroll
    for (int h = 0; h < HEADS; ++h) {
        float a1 = 0.f, a2 = 0.f;
#pragma unroll
        for (int d = 0; d < DH; ++d) {
            float s = srow[h * DH + d];
            a1 += s * wu[h * DH + d];
            a2 += s * wv[h * DH + d];
        }
        f1[(size_t)h * NN + j] = a1;
        f1t[(size_t)j * HEADS + h] = a1;
        f2[(size_t)h * NN + j] = a2;
    }
}

// ---------------- K2b: M1[g][h][i] = max over masked j of f1[h][j] ----------------
__global__ void k_maskmax(const float* __restrict__ Dg, const float* __restrict__ Sg,
                          const float* __restrict__ f1t, float* __restrict__ M1) {
    __shared__ float sm[4][32][HEADS];
    int wave = threadIdx.x >> 5, lane = threadIdx.x & 31;
    int gid = blockIdx.x * 4 + wave;             // 8192 = 2 graphs * 4096 rows
    int g = gid >> 12, row = gid & (NN - 1);
    const float* G = g ? Sg : Dg;
    float mx[HEADS];
#pragma unroll
    for (int h = 0; h < HEADS; ++h) mx[h] = -3.0e38f;
    const float* grow = G + (size_t)row * NN;
    for (int j = lane; j < NN; j += 32) {
        float gv = grow[j];
        if (gv != 0.f) {
            const float* fp = f1t + (size_t)j * HEADS;
            float4 u0 = *(const float4*)(fp);
            float4 u1 = *(const float4*)(fp + 4);
            mx[0] = fmaxf(mx[0], u0.x); mx[1] = fmaxf(mx[1], u0.y);
            mx[2] = fmaxf(mx[2], u0.z); mx[3] = fmaxf(mx[3], u0.w);
            mx[4] = fmaxf(mx[4], u1.x); mx[5] = fmaxf(mx[5], u1.y);
            mx[6] = fmaxf(mx[6], u1.z); mx[7] = fmaxf(mx[7], u1.w);
        }
    }
#pragma unroll
    for (int h = 0; h < HEADS; ++h) sm[wave][lane][h] = mx[h];
    __syncthreads();
    if (lane < HEADS) {
        float m = sm[wave][0][lane];
#pragma unroll 4
        for (int l = 1; l < 32; ++l) m = fmaxf(m, sm[wave][l][lane]);
        M1[((size_t)g * HEADS + lane) * NN + row] = m;
    }
}

// ---------------- K2c: rearrange score into f16 WMMA B-fragments per (head, j-tile) ----------------
__global__ void k_scoreB(const float* __restrict__ score, _Float16* __restrict__ scoreB) {
    int tid = blockIdx.x * blockDim.x + threadIdx.x;   // 8*128*32 = 32768 threads
    int lane = tid & 31;
    int tt = tid >> 5;
    int t = tt & (JT - 1);
    int h = tt >> 7;
    if (h >= HEADS) return;
    int jb = t * 32 + ((lane & 16) ? 8 : 0);
    int col = h * DH + (lane & 15);
    _Float16* dst = scoreB + (((size_t)(h * JT + t) * 32 + lane) << 4);
#pragma unroll
    for (int i = 0; i < 16; ++i)
        dst[i] = (_Float16)score[(size_t)(jb + (i < 8 ? i : i + 8)) * HD + col];
}

// ---------------- K3: fused masked-softmax + attn@score (per graph, head, 16-row tile) ----------------
__global__ void k_attn(const float* __restrict__ Dg, const float* __restrict__ Sg,
                       const float* __restrict__ f1, const float* __restrict__ f2,
                       const float* __restrict__ M1, const _Float16* __restrict__ scoreB,
                       float* __restrict__ supD, float* __restrict__ supS) {
    __shared__ float sden[4][32];
    int wave = threadIdx.x >> 5, lane = threadIdx.x & 31;
    int gid = blockIdx.x * 4 + wave;             // 4096 = 2 * 8 * 256
    int g = gid >> 11;
    int rem = gid & 2047;
    int h = rem >> 8;
    int it = rem & 255;
    int i0 = it * 16;
    int row = i0 + (lane & 15);

    const float* G = g ? Sg : Dg;
    float* sup = g ? supS : supD;

    float f2v = f2[(size_t)h * NN + row];
    float m1  = M1[((size_t)g * HEADS + h) * NN + row];
    float wmax = lrelu(f2v + m1);

    const float* grow = G + (size_t)row * NN;
    const float* f1h = f1 + (size_t)h * NN;
    const v16h* bbase = (const v16h*)scoreB + ((size_t)(h * JT) * 32 + lane);
    int jo = (lane & 16) ? 8 : 0;

    v8f acc = {0.f,0.f,0.f,0.f,0.f,0.f,0.f,0.f};
    float denom = 0.f;

    for (int t = 0; t < JT; ++t) {
        int jb = t * 32 + jo;
        int pj = jb + 256;                          // prefetch 8 tiles ahead
        if (pj < NN) __builtin_prefetch((const void*)(grow + pj), 0, 1);
        float4 g0 = *(const float4*)(grow + jb);
        float4 g1 = *(const float4*)(grow + jb + 4);
        float4 g2 = *(const float4*)(grow + jb + 16);
        float4 g3 = *(const float4*)(grow + jb + 20);
        float4 a0 = *(const float4*)(f1h + jb);
        float4 a1 = *(const float4*)(f1h + jb + 4);
        float4 a2 = *(const float4*)(f1h + jb + 16);
        float4 a3 = *(const float4*)(f1h + jb + 20);
        float gv[16], fv[16];
        gv[0]=g0.x; gv[1]=g0.y; gv[2]=g0.z; gv[3]=g0.w;
        gv[4]=g1.x; gv[5]=g1.y; gv[6]=g1.z; gv[7]=g1.w;
        gv[8]=g2.x; gv[9]=g2.y; gv[10]=g2.z; gv[11]=g2.w;
        gv[12]=g3.x; gv[13]=g3.y; gv[14]=g3.z; gv[15]=g3.w;
        fv[0]=a0.x; fv[1]=a0.y; fv[2]=a0.z; fv[3]=a0.w;
        fv[4]=a1.x; fv[5]=a1.y; fv[6]=a1.z; fv[7]=a1.w;
        fv[8]=a2.x; fv[9]=a2.y; fv[10]=a2.z; fv[11]=a2.w;
        fv[12]=a3.x; fv[13]=a3.y; fv[14]=a3.z; fv[15]=a3.w;

        v16h af;
#pragma unroll
        for (int i = 0; i < 16; ++i) {
            float w = lrelu(fv[i] + f2v);
            float e = __expf(w - wmax);
            e = (gv[i] != 0.f) ? e : 0.f;        // select before f16 cvt: no inf*0 NaNs
            _Float16 eh = (_Float16)e;
            af[i] = eh;
            denom += (float)eh;                   // denominator consistent with f16 numerator
        }
        v16h bf = bbase[(size_t)t * 32];
        acc = wmma_f16(af, bf, acc);
    }

    sden[wave][lane] = denom;
    __syncthreads();

    int rbase = (lane & 16) ? 8 : 0;
    int col = h * DH + (lane & 15);
#pragma unroll
    for (int r = 0; r < 8; ++r) {
        int rd = rbase + r;
        float dtot = sden[wave][rd] + sden[wave][rd + 16];
        float inv = 1.0f / fmaxf(dtot, 1e-30f);
        sup[(size_t)(i0 + rd) * HD + col] = acc[r] * inv;
    }
}

// ---------------- K4: out = S_sup@Wcat[:128] + D_sup@Wcat[128:] + inputs@proj + biases ----------------
__global__ void k_out_gemm(const float* __restrict__ supS, const float* __restrict__ supD,
                           const float* __restrict__ inputs, const _Float16* __restrict__ packC0,
                           const _Float16* __restrict__ packC1, const _Float16* __restrict__ packP,
                           const float* __restrict__ Wcat_b, const float* __restrict__ bias,
                           const float* __restrict__ proj_b, float* __restrict__ out) {
    int wave = threadIdx.x >> 5, lane = threadIdx.x & 31;
    int tile = blockIdx.x * 4 + wave;            // 2048 tiles
    if (tile >= (NN / 16) * (HD / 16)) return;
    int tm = tile >> 3, tn = tile & 7;
    int row = tm * 16 + (lane & 15);
    int col = tn * 16 + (lane & 15);
    const v16h* c0 = (const v16h*)packC0;
    const v16h* c1 = (const v16h*)packC1;
    const v16h* pp = (const v16h*)packP;
    v8f acc = {0.f,0.f,0.f,0.f,0.f,0.f,0.f,0.f};
#pragma unroll
    for (int kt = 0; kt < HD / 32; ++kt)         // S_sup x Wcat[0:128]
        acc = wmma_f16(load_afrag_f32(supS, HD, row, kt * 32, lane),
                       c0[(size_t)(kt * 8 + tn) * 32 + lane], acc);
#pragma unroll
    for (int kt = 0; kt < HD / 32; ++kt)         // D_sup x Wcat[128:256]
        acc = wmma_f16(load_afrag_f32(supD, HD, row, kt * 32, lane),
                       c1[(size_t)(kt * 8 + tn) * 32 + lane], acc);
#pragma unroll
    for (int kt = 0; kt < F_IN / 32; ++kt)       // inputs x proj
        acc = wmma_f16(load_afrag_f32(inputs, F_IN, row, kt * 32, lane),
                       pp[(size_t)(kt * 8 + tn) * 32 + lane], acc);
    float bv = Wcat_b[col] + bias[col] + proj_b[col];
    int rbase = tm * 16 + ((lane & 16) ? 8 : 0);
#pragma unroll
    for (int r = 0; r < 8; ++r)
        out[(size_t)(rbase + r) * HD + col] = acc[r] + bv;
}

extern "C" void kernel_launch(void* const* d_in, const int* in_sizes, int n_in,
                              void* d_out, int out_size, void* d_ws, size_t ws_size,
                              hipStream_t stream) {
    const float* inputs = (const float*)d_in[0];
    const float* Dg     = (const float*)d_in[1];
    const float* Sg     = (const float*)d_in[2];
    const float* W_w    = (const float*)d_in[3];
    const float* W_b    = (const float*)d_in[4];
    const float* wu     = (const float*)d_in[5];
    const float* wv     = (const float*)d_in[6];
    const float* Wcat_w = (const float*)d_in[7];
    const float* Wcat_b = (const float*)d_in[8];
    const float* bias   = (const float*)d_in[9];
    const float* proj_w = (const float*)d_in[10];
    const float* proj_b = (const float*)d_in[11];
    float* out = (float*)d_out;

    char* w = (char*)d_ws;
    float*     score  = (float*)(w + 0);                         // 2 MB
    float*     f1     = (float*)(w + 2097152);                   // 128 KB
    float*     f1t    = (float*)(w + 2228224);                   // 128 KB
    float*     f2     = (float*)(w + 2359296);                   // 128 KB
    float*     M1     = (float*)(w + 2490368);                   // 256 KB
    _Float16*  scoreB = (_Float16*)(w + 2752512);                // 1 MB
    float*     supD   = (float*)(w + 3801088);                   // 2 MB
    float*     supS   = (float*)(w + 5898240);                   // 2 MB
    _Float16*  packW  = (_Float16*)(w + 7995392);                // 64 KB
    _Float16*  packC0 = (_Float16*)(w + 8060928);                // 32 KB
    _Float16*  packC1 = (_Float16*)(w + 8093696);                // 32 KB
    _Float16*  packP  = (_Float16*)(w + 8126464);                // 64 KB

    k_packB<<<16, 128, 0, stream>>>(W_w, packW, 8);
    k_packB<<<8, 128, 0, stream>>>(Wcat_w, packC0, 4);
    k_packB<<<8, 128, 0, stream>>>(Wcat_w + (size_t)HD * HD, packC1, 4);
    k_packB<<<16, 128, 0, stream>>>(proj_w, packP, 8);
    k_score_gemm<<<512, 128, 0, stream>>>(inputs, packW, W_b, score);
    k_f1f2<<<32, 128, 0, stream>>>(score, wu, wv, f1, f1t, f2);
    k_maskmax<<<2048, 128, 0, stream>>>(Dg, Sg, f1t, M1);
    k_scoreB<<<256, 128, 0, stream>>>(score, scoreB);
    k_attn<<<1024, 128, 0, stream>>>(Dg, Sg, f1, f2, M1, scoreB, supD, supS);
    k_out_gemm<<<512, 128, 0, stream>>>(supS, supD, inputs, packC0, packC1, packP,
                                        Wcat_b, bias, proj_b, out);
}
